// DynamicSkippingMixtralSparseMoeBlockWrapper_15023795601823
// MI455X (gfx1250) — compile-verified, matrix operand order checked by reading, not run on hardware
//
#include <hip/hip_runtime.h>
#include <hip/hip_bf16.h>

// ---------------------------------------------------------------------------
// Dynamic-skipping Mixtral sparse MoE block for MI455X (gfx1250, wave32).
// Grouped-GEMM formulation with bf16 V_WMMA_F32_16X16X32_BF16 matrix cores.
//
// Pipeline (all on `stream`, graph-capture safe):
//   1. zero_counts      : reset per-expert token counters
//   2. router_k         : logits = x @ gate_w^T, softmax, top-2, beta-skip,
//                         renormalize; builds per-expert token lists via
//                         atomicAdd; writes router_logits into d_out tail
//   3. prefix_k         : exclusive prefix sum of counts -> act/y row bases
//   4. xcast_k          : x fp32 -> bf16 (single conversion, reused 2x)
//   5. gemm1_k (WMMA)   : act = silu(xg @ w1[e]^T) * (xg @ w3[e]^T)  (bf16)
//   6. gemm2_k (WMMA)   : y   = act @ w2[e]^T                       (f32)
//   7. combine_k        : out[t] = w0*y[row0] + w1*y[row1]  (gather, no
//                         float atomics -> deterministic)
//
// Workspace requirement: ~48.2 MB (see WS_* offsets below).
// ---------------------------------------------------------------------------

#define TTOK 2048
#define HD   1024
#define FD   3584
#define NEXP 8
#define BETA 0.6f

typedef __attribute__((ext_vector_type(16))) __bf16 v16bf;
typedef __attribute__((ext_vector_type(8)))  __bf16 v8bf;
typedef __attribute__((ext_vector_type(8)))  float  v8f;

union Frag  { v16bf v; v8bf h[2]; };
union AccU  { v8f   v; float f[8]; };

static __device__ __forceinline__ unsigned short f32_to_bf16(float f) {
    union { float f; unsigned u; } x; x.f = f;
    unsigned r = x.u + 0x7FFFu + ((x.u >> 16) & 1u);   // round-to-nearest-even
    return (unsigned short)(r >> 16);
}
static __device__ __forceinline__ unsigned pack2bf(float a, float b) {
    return (unsigned)f32_to_bf16(a) | ((unsigned)f32_to_bf16(b) << 16);
}
// silu with hardware v_rcp_f32 instead of IEEE division chain
static __device__ __forceinline__ float fast_silu(float v) {
    return v * __builtin_amdgcn_rcpf(1.f + __expf(-v));
}

// ---------------------------------------------------------------------------
// 1. zero per-expert counts
// ---------------------------------------------------------------------------
__global__ void zero_counts_k(int* counts) {
    if (threadIdx.x < NEXP) counts[threadIdx.x] = 0;
}

// ---------------------------------------------------------------------------
// 2. router: one wave (32 lanes) per token
// ---------------------------------------------------------------------------
__global__ __launch_bounds__(32)
void router_k(const float* __restrict__ x, const float* __restrict__ gw,
              float* __restrict__ logits_out, int* __restrict__ counts,
              int* __restrict__ mi, float* __restrict__ mw,
              int* __restrict__ tok_idx) {
    const int t = blockIdx.x;
    const int lane = threadIdx.x;
    const float* xr = x + (size_t)t * HD;

    float acc[NEXP];
#pragma unroll
    for (int e = 0; e < NEXP; ++e) acc[e] = 0.f;

    for (int h = lane; h < HD; h += 32) {
        const float xv = xr[h];
#pragma unroll
        for (int e = 0; e < NEXP; ++e) acc[e] += xv * gw[e * HD + h];
    }
#pragma unroll
    for (int off = 16; off > 0; off >>= 1) {
#pragma unroll
        for (int e = 0; e < NEXP; ++e) acc[e] += __shfl_xor(acc[e], off, 32);
    }

    if (lane == 0) {
        // raw logits out (second tuple output)
#pragma unroll
        for (int e = 0; e < NEXP; ++e) logits_out[(size_t)t * NEXP + e] = acc[e];

        // softmax
        float m = acc[0];
#pragma unroll
        for (int e = 1; e < NEXP; ++e) m = fmaxf(m, acc[e]);
        float p[NEXP], s = 0.f;
#pragma unroll
        for (int e = 0; e < NEXP; ++e) { p[e] = __expf(acc[e] - m); s += p[e]; }
        const float inv = 1.f / s;
#pragma unroll
        for (int e = 0; e < NEXP; ++e) p[e] *= inv;

        // top-2 (lowest index wins ties, matching lax.top_k)
        int i0 = 0;
#pragma unroll
        for (int e = 1; e < NEXP; ++e) if (p[e] > p[i0]) i0 = e;
        int i1 = (i0 == 0) ? 1 : 0;
#pragma unroll
        for (int e = 0; e < NEXP; ++e)
            if (e != i0 && p[e] > p[i1]) i1 = e;

        const float p0 = p[i0], p1 = p[i1];
        const bool skip = (p1 < BETA * p0);
        float w0, w1;
        if (skip) { w0 = 1.f; w1 = 0.f; }
        else { const float d = 1.f / (p0 + p1); w0 = p0 * d; w1 = p1 * d; }

        const int s0 = atomicAdd(&counts[i0], 1);
        tok_idx[i0 * TTOK + s0] = t;
        mi[t * 4 + 0] = i0; mi[t * 4 + 1] = s0;
        mw[t * 2 + 0] = w0;
        if (!skip) {
            const int s1 = atomicAdd(&counts[i1], 1);
            tok_idx[i1 * TTOK + s1] = t;
            mi[t * 4 + 2] = i1; mi[t * 4 + 3] = s1;
            mw[t * 2 + 1] = w1;
        } else {
            mi[t * 4 + 2] = -1; mi[t * 4 + 3] = 0;
            mw[t * 2 + 1] = 0.f;
        }
    }
}

// ---------------------------------------------------------------------------
// 3. exclusive prefix sum of per-expert counts
// ---------------------------------------------------------------------------
__global__ void prefix_k(const int* __restrict__ counts, int* __restrict__ base) {
    if (threadIdx.x == 0) {
        int s = 0;
#pragma unroll
        for (int e = 0; e < NEXP; ++e) { base[e] = s; s += counts[e]; }
    }
}

// ---------------------------------------------------------------------------
// 4. x fp32 -> bf16 (8 elements / thread)
// ---------------------------------------------------------------------------
__global__ __launch_bounds__(256)
void xcast_k(const float* __restrict__ x, unsigned short* __restrict__ xb) {
    const size_t i = ((size_t)blockIdx.x * blockDim.x + threadIdx.x) * 8;
    const float4 a = ((const float4*)(x + i))[0];
    const float4 b = ((const float4*)(x + i))[1];
    uint4 pkt;
    pkt.x = pack2bf(a.x, a.y); pkt.y = pack2bf(a.z, a.w);
    pkt.z = pack2bf(b.x, b.y); pkt.w = pack2bf(b.z, b.w);
    *(uint4*)(xb + i) = pkt;
}

// ---------------------------------------------------------------------------
// 5. GEMM1: act = silu(xg @ w1^T) * (xg @ w3^T), per expert, gathered rows.
//    Block = 64(M) x 64(N) tile, 8 waves; wave -> 16x32 subtile; K-step 32.
//    Weights streamed fp32 -> bf16 during LDS staging; next k-step weight
//    lines prefetched (global_prefetch_b8) to overlap HBM with WMMA.
// ---------------------------------------------------------------------------
__global__ __launch_bounds__(256)
void gemm1_k(const unsigned short* __restrict__ xb,
             const float* __restrict__ w1g, const float* __restrict__ w3g,
             const int* __restrict__ counts, const int* __restrict__ base,
             const int* __restrict__ tok_idx,
             unsigned short* __restrict__ act) {
    const int e = blockIdx.z;
    const int count = counts[e];
    const int m0 = blockIdx.x * 64;
    if (m0 >= count) return;
    const int n0 = blockIdx.y * 64;
    const int abase = base[e];

    __shared__ unsigned short lA[64 * 32];
    __shared__ unsigned short lB1[64 * 32];
    __shared__ unsigned short lB3[64 * 32];

    const int tid = threadIdx.x;
    const int lane = tid & 31, wave = tid >> 5;
    const int wm = wave >> 1, wn = wave & 1;          // 4 M-rows x 2 N-cols
    const int lrow = tid >> 2, lcol = (tid & 3) * 8;  // tile staging

    const int gm = m0 + lrow;
    const int tok = (gm < count) ? tok_idx[e * TTOK + gm] : -1;
    const unsigned short* xrow = xb + (size_t)(tok < 0 ? 0 : tok) * HD + lcol;
    const float* w1r = w1g + (size_t)e * FD * HD + (size_t)(n0 + lrow) * HD + lcol;
    const float* w3r = w3g + (size_t)e * FD * HD + (size_t)(n0 + lrow) * HD + lcol;

    AccU c1a, c1b, c3a, c3b;
#pragma unroll
    for (int i = 0; i < 8; ++i) { c1a.f[i] = 0; c1b.f[i] = 0; c3a.f[i] = 0; c3b.f[i] = 0; }

    const int kA = (lane >> 4) * 8;          // ISA 16-bit operand lane split
    const int lo = lane & 15;

    for (int k0 = 0; k0 < HD; k0 += 32) {
        // stage A (gathered bf16 tokens)
        uint4 av = make_uint4(0u, 0u, 0u, 0u);
        if (tok >= 0) av = *(const uint4*)(xrow + k0);
        *(uint4*)&lA[lrow * 32 + lcol] = av;
        // stage B1/B3 (fp32 weights -> bf16)
        {
            const float4 f0 = ((const float4*)(w1r + k0))[0];
            const float4 f1 = ((const float4*)(w1r + k0))[1];
            uint4 p; p.x = pack2bf(f0.x, f0.y); p.y = pack2bf(f0.z, f0.w);
            p.z = pack2bf(f1.x, f1.y); p.w = pack2bf(f1.z, f1.w);
            *(uint4*)&lB1[lrow * 32 + lcol] = p;
        }
        {
            const float4 f0 = ((const float4*)(w3r + k0))[0];
            const float4 f1 = ((const float4*)(w3r + k0))[1];
            uint4 p; p.x = pack2bf(f0.x, f0.y); p.y = pack2bf(f0.z, f0.w);
            p.z = pack2bf(f1.x, f1.y); p.w = pack2bf(f1.z, f1.w);
            *(uint4*)&lB3[lrow * 32 + lcol] = p;
        }
        // prefetch next k-step weight lines (overlaps HBM with WMMA below)
        if (k0 + 32 < HD) {
            __builtin_prefetch(w1r + k0 + 32, 0, 1);
            __builtin_prefetch(w3r + k0 + 32, 0, 1);
        }
        __syncthreads();

        Frag a;
        const int ar = wm * 16 + lo;
        a.h[0] = *(const v8bf*)&lA[ar * 32 + kA];
        a.h[1] = *(const v8bf*)&lA[ar * 32 + kA + 16];

        const int bc0 = wn * 32 + lo;
        Frag b;
        b.h[0] = *(const v8bf*)&lB1[bc0 * 32 + kA];
        b.h[1] = *(const v8bf*)&lB1[bc0 * 32 + kA + 16];
        c1a.v = __builtin_amdgcn_wmma_f32_16x16x32_bf16(false, a.v, false, b.v,
                                                        (short)0, c1a.v, false, false);
        b.h[0] = *(const v8bf*)&lB1[(bc0 + 16) * 32 + kA];
        b.h[1] = *(const v8bf*)&lB1[(bc0 + 16) * 32 + kA + 16];
        c1b.v = __builtin_amdgcn_wmma_f32_16x16x32_bf16(false, a.v, false, b.v,
                                                        (short)0, c1b.v, false, false);
        b.h[0] = *(const v8bf*)&lB3[bc0 * 32 + kA];
        b.h[1] = *(const v8bf*)&lB3[bc0 * 32 + kA + 16];
        c3a.v = __builtin_amdgcn_wmma_f32_16x16x32_bf16(false, a.v, false, b.v,
                                                        (short)0, c3a.v, false, false);
        b.h[0] = *(const v8bf*)&lB3[(bc0 + 16) * 32 + kA];
        b.h[1] = *(const v8bf*)&lB3[(bc0 + 16) * 32 + kA + 16];
        c3b.v = __builtin_amdgcn_wmma_f32_16x16x32_bf16(false, a.v, false, b.v,
                                                        (short)0, c3b.v, false, false);
        __syncthreads();
    }

    // epilogue: silu(c1) * c3 -> act (bf16)
    const int mo = (lane >= 16) ? 8 : 0;
#pragma unroll
    for (int r = 0; r < 8; ++r) {
        const int gmr = m0 + wm * 16 + mo + r;
        if (gmr < count) {
            const size_t arow = (size_t)(abase + gmr) * FD;
            act[arow + n0 + wn * 32 + lo]      = f32_to_bf16(fast_silu(c1a.f[r]) * c3a.f[r]);
            act[arow + n0 + wn * 32 + 16 + lo] = f32_to_bf16(fast_silu(c1b.f[r]) * c3b.f[r]);
        }
    }
}

// ---------------------------------------------------------------------------
// 6. GEMM2: y = act @ w2^T, per expert.  Same tiling, K over F (112 steps).
// ---------------------------------------------------------------------------
__global__ __launch_bounds__(256)
void gemm2_k(const unsigned short* __restrict__ act,
             const float* __restrict__ w2g,
             const int* __restrict__ counts, const int* __restrict__ base,
             float* __restrict__ y) {
    const int e = blockIdx.z;
    const int count = counts[e];
    const int m0 = blockIdx.x * 64;
    if (m0 >= count) return;
    const int n0 = blockIdx.y * 64;
    const int abase = base[e];

    __shared__ unsigned short lA[64 * 32];
    __shared__ unsigned short lB[64 * 32];

    const int tid = threadIdx.x;
    const int lane = tid & 31, wave = tid >> 5;
    const int wm = wave >> 1, wn = wave & 1;
    const int lrow = tid >> 2, lcol = (tid & 3) * 8;

    const int gm = m0 + lrow;
    const unsigned short* arow_p =
        act + (size_t)(abase + (gm < count ? gm : 0)) * FD + lcol;
    const float* w2r = w2g + (size_t)e * HD * FD + (size_t)(n0 + lrow) * FD + lcol;

    AccU ca, cb;
#pragma unroll
    for (int i = 0; i < 8; ++i) { ca.f[i] = 0; cb.f[i] = 0; }

    const int kA = (lane >> 4) * 8;
    const int lo = lane & 15;

    for (int k0 = 0; k0 < FD; k0 += 32) {
        uint4 av = make_uint4(0u, 0u, 0u, 0u);
        if (gm < count) av = *(const uint4*)(arow_p + k0);
        *(uint4*)&lA[lrow * 32 + lcol] = av;
        {
            const float4 f0 = ((const float4*)(w2r + k0))[0];
            const float4 f1 = ((const float4*)(w2r + k0))[1];
            uint4 p; p.x = pack2bf(f0.x, f0.y); p.y = pack2bf(f0.z, f0.w);
            p.z = pack2bf(f1.x, f1.y); p.w = pack2bf(f1.z, f1.w);
            *(uint4*)&lB[lrow * 32 + lcol] = p;
        }
        if (k0 + 32 < FD) {
            __builtin_prefetch(w2r + k0 + 32, 0, 1);
            __builtin_prefetch(arow_p + k0 + 32, 0, 1);
        }
        __syncthreads();

        Frag a;
        const int ar = wm * 16 + lo;
        a.h[0] = *(const v8bf*)&lA[ar * 32 + kA];
        a.h[1] = *(const v8bf*)&lA[ar * 32 + kA + 16];

        const int bc0 = wn * 32 + lo;
        Frag b;
        b.h[0] = *(const v8bf*)&lB[bc0 * 32 + kA];
        b.h[1] = *(const v8bf*)&lB[bc0 * 32 + kA + 16];
        ca.v = __builtin_amdgcn_wmma_f32_16x16x32_bf16(false, a.v, false, b.v,
                                                       (short)0, ca.v, false, false);
        b.h[0] = *(const v8bf*)&lB[(bc0 + 16) * 32 + kA];
        b.h[1] = *(const v8bf*)&lB[(bc0 + 16) * 32 + kA + 16];
        cb.v = __builtin_amdgcn_wmma_f32_16x16x32_bf16(false, a.v, false, b.v,
                                                       (short)0, cb.v, false, false);
        __syncthreads();
    }

    const int mo = (lane >= 16) ? 8 : 0;
#pragma unroll
    for (int r = 0; r < 8; ++r) {
        const int gmr = m0 + wm * 16 + mo + r;
        if (gmr < count) {
            const size_t yrow = (size_t)(abase + gmr) * HD;
            y[yrow + n0 + wn * 32 + lo]      = ca.f[r];
            y[yrow + n0 + wn * 32 + 16 + lo] = cb.f[r];
        }
    }
}

// ---------------------------------------------------------------------------
// 7. combine: out[t] = w0*y[row0] + w1*y[row1]  (deterministic gather)
// ---------------------------------------------------------------------------
__global__ __launch_bounds__(256)
void combine_k(const float* __restrict__ y, const int* __restrict__ mi,
               const float* __restrict__ mw, const int* __restrict__ base,
               float* __restrict__ out) {
    const int t = blockIdx.x, tid = threadIdx.x;
    const int e0 = mi[t * 4 + 0], s0 = mi[t * 4 + 1];
    const int e1 = mi[t * 4 + 2], s1 = mi[t * 4 + 3];
    const float w0 = mw[t * 2 + 0], w1v = mw[t * 2 + 1];

    const float4 a = ((const float4*)(y + (size_t)(base[e0] + s0) * HD))[tid];
    float4 r;
    r.x = w0 * a.x; r.y = w0 * a.y; r.z = w0 * a.z; r.w = w0 * a.w;
    if (e1 >= 0) {
        const float4 b = ((const float4*)(y + (size_t)(base[e1] + s1) * HD))[tid];
        r.x += w1v * b.x; r.y += w1v * b.y; r.z += w1v * b.z; r.w += w1v * b.w;
    }
    ((float4*)(out + (size_t)t * HD))[tid] = r;
}

// ---------------------------------------------------------------------------
// workspace layout (bytes, all 256-aligned); total ~48.2 MB
// ---------------------------------------------------------------------------
#define WS_COUNTS 0
#define WS_BASE   256
#define WS_MI     512                                   // T*4 ints  = 32 KB
#define WS_MW     (WS_MI + TTOK * 4 * 4)                // T*2 f32   = 16 KB
#define WS_TOKIDX (WS_MW + TTOK * 2 * 4)                // E*T ints  = 64 KB
#define WS_XBF    (WS_TOKIDX + NEXP * TTOK * 4)         // T*H bf16  = 4 MB
#define WS_ACT    (WS_XBF + (size_t)TTOK * HD * 2)      // 4096*F bf16 = 28 MB
#define WS_Y      (WS_ACT + (size_t)2 * TTOK * FD * 2)  // 4096*H f32  = 16 MB

extern "C" void kernel_launch(void* const* d_in, const int* in_sizes, int n_in,
                              void* d_out, int out_size, void* d_ws, size_t ws_size,
                              hipStream_t stream) {
    (void)in_sizes; (void)n_in; (void)out_size; (void)ws_size;

    const float* x   = (const float*)d_in[0];   // [T, H]
    const float* gw  = (const float*)d_in[1];   // [E, H]
    const float* w1g = (const float*)d_in[2];   // [E, F, H]
    const float* w3g = (const float*)d_in[3];   // [E, F, H]
    const float* w2g = (const float*)d_in[4];   // [E, H, F]

    float* out        = (float*)d_out;                       // [T, H]
    float* logits_out = (float*)d_out + (size_t)TTOK * HD;   // [T, E]

    char* ws = (char*)d_ws;
    int*            counts  = (int*)(ws + WS_COUNTS);
    int*            basep   = (int*)(ws + WS_BASE);
    int*            mi      = (int*)(ws + WS_MI);
    float*          mw      = (float*)(ws + WS_MW);
    int*            tok_idx = (int*)(ws + WS_TOKIDX);
    unsigned short* xb      = (unsigned short*)(ws + WS_XBF);
    unsigned short* act     = (unsigned short*)(ws + WS_ACT);
    float*          yb      = (float*)(ws + WS_Y);

    zero_counts_k<<<1, 32, 0, stream>>>(counts);
    router_k<<<TTOK, 32, 0, stream>>>(x, gw, logits_out, counts, mi, mw, tok_idx);
    prefix_k<<<1, 32, 0, stream>>>(counts, basep);
    xcast_k<<<(TTOK * HD) / (256 * 8), 256, 0, stream>>>(x, xb);

    dim3 g1((TTOK + 63) / 64, FD / 64, NEXP);   // (32, 56, 8)
    gemm1_k<<<g1, 256, 0, stream>>>(xb, w1g, w3g, counts, basep, tok_idx, act);

    dim3 g2((TTOK + 63) / 64, HD / 64, NEXP);   // (32, 16, 8)
    gemm2_k<<<g2, 256, 0, stream>>>(act, w2g, counts, basep, yb);

    combine_k<<<TTOK, 256, 0, stream>>>(yb, mi, mw, basep, out);
}
